// GNNModel_71236327571596
// MI455X (gfx1250) — compile-verified, matrix operand order checked by reading, not run on hardware
//
#include <hip/hip_runtime.h>
#include <hip/hip_bf16.h>
#include <math.h>

// ---------------------------------------------------------------------------
// GNN forward for MI455X (gfx1250, wave32).
// Dense GEMMs: v_wmma_f32_16x16x32_bf16, 16x64 output strip per wave (NT=4),
// bf16 operands pre-staged (X converted once, W transposed to K-major bf16)
// so fragments are contiguous b128 loads and no cvt sits in the WMMA loop.
// Graph aggregation uses float atomicAdd scatter (L2-resident at 192MB).
// ---------------------------------------------------------------------------

typedef __attribute__((ext_vector_type(16))) __bf16      v16bf;
typedef __attribute__((ext_vector_type(8)))  float       v8f;
typedef __attribute__((ext_vector_type(4)))  unsigned    u32x4;  // 16B, align 16

#define TPB 256

__device__ __forceinline__ float lrelu(float x, float s) { return x > 0.0f ? x : s * x; }

// monotone float<->uint mapping so atomicMax(uint) orders floats correctly
__device__ __forceinline__ unsigned fflip(float f) {
  unsigned u = __float_as_uint(f);
  return (u & 0x80000000u) ? ~u : (u | 0x80000000u);
}
__device__ __forceinline__ float funflip(unsigned u) {
  u = (u & 0x80000000u) ? (u ^ 0x80000000u) : ~u;
  return __uint_as_float(u);
}

// ---------------- generic helpers ----------------
__global__ void k_fill(float* __restrict__ p, size_t n, float v) {
  size_t t = (size_t)blockIdx.x * blockDim.x + threadIdx.x;
  if (t < n) p[t] = v;
}

__global__ void k_deg(const int* __restrict__ dst, float* __restrict__ deg, long long E) {
  long long t = (long long)blockIdx.x * blockDim.x + threadIdx.x;
  if (t < E) atomicAdd(&deg[dst[t]], 1.0f);
}

__global__ void k_rsqrt_inplace(float* __restrict__ p, int n) {
  int t = blockIdx.x * blockDim.x + threadIdx.x;
  if (t < n) p[t] = rsqrtf(p[t]);
}

// ---------------- bf16 staging ----------------
__global__ void f32_to_bf16(const float* __restrict__ X, __bf16* __restrict__ Y, size_t n) {
  size_t t = (size_t)blockIdx.x * blockDim.x + threadIdx.x;
  if (t < n) Y[t] = (__bf16)X[t];
}

// Wt[n*K + k] = bf16(W[k*F + n])  (weights tiny: <= 512x256)
__global__ void w_transpose_bf16(const float* __restrict__ W, __bf16* __restrict__ Wt,
                                 int K, int F) {
  int t = blockIdx.x * blockDim.x + threadIdx.x;
  if (t >= K * F) return;
  const int k = t / F, n = t % F;
  Wt[(size_t)n * K + k] = (__bf16)W[t];
}

// ---------------- WMMA GEMM: Y[N,F] = X[N,K] @ W[K,F] ----------------------
// Xb: bf16 row-major [N,K]; Wt: bf16 K-major [F,K]. One 16x(16*NT) strip per
// wave: A fragment loaded once per K-step, reused over NT B fragments.
// Requires N%16==0, K%32==0, F%(16*NT)==0 (all shapes here comply).
template <int NT>
__global__ void __launch_bounds__(TPB)
gemm_wmma_bf16(const __bf16* __restrict__ Xb, const __bf16* __restrict__ Wt,
               float* __restrict__ Y, int nRows, int K, int Fout) {
  const int wave = threadIdx.x >> 5;
  const int lane = threadIdx.x & 31;
  const int colGroups = Fout / (16 * NT);
  const int tiles = (nRows >> 4) * colGroups;
  const int tile = blockIdx.x * (TPB >> 5) + wave;
  if (tile >= tiles) return;          // wave-uniform: EXEC all-ones inside

  const int row0 = (tile / colGroups) << 4;
  const int col0 = (tile % colGroups) * (16 * NT);
  const int m  = row0 + (lane & 15);  // A: lane holds row m
  const int kh = (lane >> 4) << 3;    // K half-group base: 0 or 8

  const __bf16* xr = Xb + (size_t)m * K;
  const __bf16* wr[NT];
#pragma unroll
  for (int t = 0; t < NT; ++t)
    wr[t] = Wt + (size_t)(col0 + t * 16 + (lane & 15)) * K;

  v8f c[NT] = {};
  for (int k0 = 0; k0 < K; k0 += 32) {
    v16bf a;
    ((u32x4*)&a)[0] = *(const u32x4*)(xr + k0 + kh);       // K = k0+kh..+7
    ((u32x4*)&a)[1] = *(const u32x4*)(xr + k0 + 16 + kh);  // K = k0+16+kh..+7
#pragma unroll
    for (int t = 0; t < NT; ++t) {
      v16bf b;
      ((u32x4*)&b)[0] = *(const u32x4*)(wr[t] + k0 + kh);
      ((u32x4*)&b)[1] = *(const u32x4*)(wr[t] + k0 + 16 + kh);
      c[t] = __builtin_amdgcn_wmma_f32_16x16x32_bf16(
          false, a, false, b, (short)0, c[t], false, false);
    }
  }
  // D layout: lanes 0-15 -> M=row0+r, lanes 16-31 -> M=row0+8+r; N = lane%16
  const int mBase = row0 + ((lane >> 4) << 3);
#pragma unroll
  for (int t = 0; t < NT; ++t) {
    const int n = col0 + t * 16 + (lane & 15);
#pragma unroll
    for (int r = 0; r < 8; ++r)
      Y[(size_t)(mBase + r) * Fout + n] = c[t][r];
  }
}

// ---------------- GCN aggregation ----------------
__global__ void gcn_scatter(const float* __restrict__ H, float* __restrict__ Agg,
                            const int* __restrict__ src, const int* __restrict__ dst,
                            const float* __restrict__ dinv, long long E, int F) {
  long long t = (long long)blockIdx.x * blockDim.x + threadIdx.x;
  if (t >= E * (long long)F) return;
  const int e = (int)(t / F), f = (int)(t % F);
  const int s = src[e], d = dst[e];
  const float coef = dinv[s] * dinv[d];
  atomicAdd(&Agg[(size_t)d * F + f], H[(size_t)s * F + f] * coef);
}

__global__ void gcn_finalize(float* __restrict__ Agg, const float* __restrict__ H,
                             const float* __restrict__ dinv, const float* __restrict__ bias,
                             size_t total, int F) {
  size_t t = (size_t)blockIdx.x * blockDim.x + threadIdx.x;
  if (t >= total) return;
  const int f = (int)(t % F);
  const float di = dinv[t / F];
  Agg[t] = Agg[t] + H[t] * di * di + bias[f];
}

// ---------------- BatchNorm (training-mode batch stats) + leaky relu -------
__global__ void colstats(const float* __restrict__ Y, float* __restrict__ stats,
                         int nRows, int F) {
  const int f = blockIdx.x;
  float s = 0.0f, s2 = 0.0f;
  for (int i = threadIdx.x; i < nRows; i += blockDim.x) {
    const float v = Y[(size_t)i * F + f];
    s += v; s2 += v * v;
  }
  __shared__ float sh[TPB], sh2[TPB];
  sh[threadIdx.x] = s; sh2[threadIdx.x] = s2; __syncthreads();
  for (int st = TPB / 2; st > 0; st >>= 1) {
    if (threadIdx.x < st) { sh[threadIdx.x] += sh[threadIdx.x + st];
                            sh2[threadIdx.x] += sh2[threadIdx.x + st]; }
    __syncthreads();
  }
  if (threadIdx.x == 0) {
    const float mean = sh[0] / nRows;
    stats[f]       = mean;
    stats[512 + f] = sh2[0] / nRows - mean * mean;  // population var (ddof=0)
  }
}

__global__ void bn_act(float* __restrict__ Y, const float* __restrict__ stats,
                       const float* __restrict__ g, const float* __restrict__ be,
                       size_t total, int F) {
  size_t t = (size_t)blockIdx.x * blockDim.x + threadIdx.x;
  if (t >= total) return;
  const int f = (int)(t % F);
  const float x = (Y[t] - stats[f]) * rsqrtf(stats[512 + f] + 1e-5f) * g[f] + be[f];
  Y[t] = lrelu(x, 0.01f);
}

// ---------------- GAT (4 heads x 128, concat=False) ----------------
__global__ void gat_attn_coef(const float* __restrict__ H,
                              const float* __restrict__ a_src, const float* __restrict__ a_dst,
                              float* __restrict__ asc, float* __restrict__ adc,
                              float* __restrict__ eself, unsigned* __restrict__ mxu, int N) {
  int t = blockIdx.x * blockDim.x + threadIdx.x;
  if (t >= N * 4) return;
  const int n = t >> 2, h = t & 3;
  const float* hp = H + (size_t)n * 512 + h * 128;
  float s = 0.0f, d = 0.0f;
  for (int f = 0; f < 128; ++f) {
    s += hp[f] * a_src[h * 128 + f];
    d += hp[f] * a_dst[h * 128 + f];
  }
  asc[t] = s; adc[t] = d;
  const float eself_raw = lrelu(s + d, 0.2f);   // self-loop logit
  eself[t] = eself_raw;
  mxu[t] = fflip(eself_raw);                    // init running max with self
}

__global__ void gat_edge_max(const int* __restrict__ src, const int* __restrict__ dst,
                             const float* __restrict__ asc, const float* __restrict__ adc,
                             unsigned* __restrict__ mxu, long long E) {
  long long t = (long long)blockIdx.x * blockDim.x + threadIdx.x;
  if (t >= E * 4) return;
  const int e = (int)(t >> 2), h = (int)(t & 3);
  const float v = lrelu(asc[src[e] * 4 + h] + adc[dst[e] * 4 + h], 0.2f);
  atomicMax(&mxu[dst[e] * 4 + h], fflip(v));
}

__global__ void gat_node_mid(unsigned* __restrict__ mxu, float* __restrict__ mxf,
                             float* __restrict__ eself, float* __restrict__ denom, int N) {
  int t = blockIdx.x * blockDim.x + threadIdx.x;
  if (t >= N * 4) return;
  const float m = funflip(mxu[t]);
  mxf[t] = m;                        // convert max to plain float, in place
  const float es = __expf(eself[t] - m);
  eself[t] = es;                     // now holds exp(self - max)
  denom[t] = es;                     // start denominator with self term
}

__global__ void gat_edge_exp(const int* __restrict__ src, const int* __restrict__ dst,
                             const float* __restrict__ asc, const float* __restrict__ adc,
                             const float* __restrict__ mxf, float* __restrict__ ee,
                             float* __restrict__ denom, long long E) {
  long long t = (long long)blockIdx.x * blockDim.x + threadIdx.x;
  if (t >= E * 4) return;
  const int e = (int)(t >> 2), h = (int)(t & 3);
  const int d = dst[e];
  const float v = lrelu(asc[src[e] * 4 + h] + adc[d * 4 + h], 0.2f);
  const float x = __expf(v - mxf[d * 4 + h]);
  ee[t] = x;
  atomicAdd(&denom[d * 4 + h], x);
}

__global__ void gat_edge_scatter(const int* __restrict__ src, const int* __restrict__ dst,
                                 const float* __restrict__ H, const float* __restrict__ ee,
                                 float* __restrict__ Acc, long long E) {
  long long t = (long long)blockIdx.x * blockDim.x + threadIdx.x;
  if (t >= E * 512) return;
  const int e = (int)(t >> 9), hf = (int)(t & 511), h = hf >> 7;
  atomicAdd(&Acc[(size_t)dst[e] * 512 + hf],
            ee[e * 4 + h] * H[(size_t)src[e] * 512 + hf]);
}

__global__ void gat_finalize(const float* __restrict__ Acc, const float* __restrict__ H,
                             const float* __restrict__ eself, const float* __restrict__ denom,
                             const float* __restrict__ bias, float* __restrict__ Out, int N) {
  int t = blockIdx.x * blockDim.x + threadIdx.x;
  if (t >= N * 128) return;
  const int n = t >> 7, f = t & 127;
  float s = 0.0f;
#pragma unroll
  for (int h = 0; h < 4; ++h) {
    const size_t idx = (size_t)n * 512 + h * 128 + f;
    s += (Acc[idx] + eself[n * 4 + h] * H[idx]) / (denom[n * 4 + h] + 1e-16f);
  }
  Out[t] = s * 0.25f + bias[f];
}

// ---------------- MLP head + log_softmax ----------------
__global__ void head_kernel(const float* __restrict__ Y5,
                            const float* __restrict__ fw1, const float* __restrict__ fb1,
                            const float* __restrict__ fw2, const float* __restrict__ fb2,
                            float* __restrict__ out, int N) {
  int n = blockIdx.x * blockDim.x + threadIdx.x;
  if (n >= N) return;
  float x[16];
#pragma unroll
  for (int i = 0; i < 16; ++i) x[i] = Y5[(size_t)n * 16 + i];
  float z1[8];
#pragma unroll
  for (int j = 0; j < 8; ++j) {
    float s = fb1[j];
#pragma unroll
    for (int i = 0; i < 16; ++i) s += x[i] * fw1[i * 8 + j];
    z1[j] = lrelu(s, 0.01f);
  }
  float z2[3];
#pragma unroll
  for (int k = 0; k < 3; ++k) {
    float s = fb2[k];
#pragma unroll
    for (int j = 0; j < 8; ++j) s += z1[j] * fw2[j * 3 + k];
    z2[k] = s;
  }
  float m = fmaxf(z2[0], fmaxf(z2[1], z2[2]));
  float l = logf(__expf(z2[0] - m) + __expf(z2[1] - m) + __expf(z2[2] - m));
#pragma unroll
  for (int k = 0; k < 3; ++k) out[(size_t)n * 3 + k] = z2[k] - m - l;
}

// ---------------------------------------------------------------------------
#define CDIV(a, b) (((a) + (b) - 1) / (b))

static inline void launch_gemm(const float* X, const float* W, float* Y,
                               int N, int K, int F,
                               __bf16* Xb, __bf16* Wt, hipStream_t s) {
  f32_to_bf16<<<CDIV((size_t)N * K, TPB), TPB, 0, s>>>(X, Xb, (size_t)N * K);
  w_transpose_bf16<<<CDIV(K * F, TPB), TPB, 0, s>>>(W, Wt, K, F);
  if (F % 64 == 0) {
    const int tiles = (N / 16) * (F / 64);
    gemm_wmma_bf16<4><<<CDIV(tiles, TPB / 32), TPB, 0, s>>>(Xb, Wt, Y, N, K, F);
  } else if (F % 32 == 0) {
    const int tiles = (N / 16) * (F / 32);
    gemm_wmma_bf16<2><<<CDIV(tiles, TPB / 32), TPB, 0, s>>>(Xb, Wt, Y, N, K, F);
  } else {
    const int tiles = (N / 16) * (F / 16);
    gemm_wmma_bf16<1><<<CDIV(tiles, TPB / 32), TPB, 0, s>>>(Xb, Wt, Y, N, K, F);
  }
}

static inline void run_bn_act(float* Y, float* stats, const float* g, const float* be,
                              int N, int F, hipStream_t s) {
  colstats<<<F, TPB, 0, s>>>(Y, stats, N, F);
  bn_act<<<CDIV((size_t)N * F, TPB), TPB, 0, s>>>(Y, stats, g, be, (size_t)N * F, F);
}

extern "C" void kernel_launch(void* const* d_in, const int* in_sizes, int n_in,
                              void* d_out, int out_size, void* d_ws, size_t ws_size,
                              hipStream_t stream) {
  const float* x      = (const float*)d_in[0];
  const int*   ei     = (const int*)d_in[1];
  const float* w1     = (const float*)d_in[2];
  const float* b1     = (const float*)d_in[3];
  const float* g1     = (const float*)d_in[4];
  const float* be1    = (const float*)d_in[5];
  const float* gat_w  = (const float*)d_in[6];
  const float* gat_as = (const float*)d_in[7];
  const float* gat_ad = (const float*)d_in[8];
  const float* gat_b  = (const float*)d_in[9];
  const float* g2     = (const float*)d_in[10];
  const float* be2    = (const float*)d_in[11];
  const float* w3     = (const float*)d_in[12];
  const float* b3     = (const float*)d_in[13];
  const float* g3     = (const float*)d_in[14];
  const float* be3    = (const float*)d_in[15];
  const float* w4     = (const float*)d_in[16];
  const float* b4     = (const float*)d_in[17];
  const float* g4     = (const float*)d_in[18];
  const float* be4    = (const float*)d_in[19];
  const float* w5     = (const float*)d_in[20];
  const float* b5     = (const float*)d_in[21];
  const float* g5     = (const float*)d_in[22];
  const float* be5    = (const float*)d_in[23];
  const float* fw1    = (const float*)d_in[24];
  const float* fb1    = (const float*)d_in[25];
  const float* fw2    = (const float*)d_in[26];
  const float* fb2    = (const float*)d_in[27];

  const int       N = in_sizes[0] / 256;        // 50000
  const long long E = in_sizes[1] / 2;          // 400000
  const int* src = ei;
  const int* dst = ei + E;

  // workspace carve-up (floats; all offsets stay 16B aligned)
  float* ws = (float*)d_ws;
  size_t off = 0;
  float* A     = ws + off; off += (size_t)N * 512;   // gemm output / GAT h
  float* B     = ws + off; off += (size_t)N * 512;   // agg / GAT acc / ping
  float* C     = ws + off; off += (size_t)N * 256;   // pong
  float* dinv  = ws + off; off += (size_t)N;
  float* asc   = ws + off; off += (size_t)N * 4;
  float* adc   = ws + off; off += (size_t)N * 4;
  float* mxf   = ws + off; off += (size_t)N * 4;     // aliased uint/float
  unsigned* mxu = (unsigned*)mxf;
  float* denom = ws + off; off += (size_t)N * 4;
  float* eself = ws + off; off += (size_t)N * 4;
  float* ee    = ws + off; off += (size_t)E * 4;
  float* stats = ws + off; off += 1024;
  __bf16* Xb   = (__bf16*)(ws + off); off += (size_t)N * 128;  // N*256 bf16 (max K=256)
  __bf16* Wt   = (__bf16*)(ws + off); off += 65536;            // 512*256 bf16 max
  (void)ws_size; (void)n_in; (void)out_size;

  // ---- degrees: deg = 1 + in-degree; dinv = rsqrt(deg) ----
  k_fill<<<CDIV((size_t)N, TPB), TPB, 0, stream>>>(dinv, N, 1.0f);
  k_deg<<<CDIV(E, (long long)TPB), TPB, 0, stream>>>(dst, dinv, E);
  k_rsqrt_inplace<<<CDIV(N, TPB), TPB, 0, stream>>>(dinv, N);

  // ---- Layer 1: GCN 256 -> 256 ----
  launch_gemm(x, w1, A, N, 256, 256, Xb, Wt, stream);
  k_fill<<<CDIV((size_t)N * 256, TPB), TPB, 0, stream>>>(B, (size_t)N * 256, 0.0f);
  gcn_scatter<<<CDIV(E * 256, (long long)TPB), TPB, 0, stream>>>(A, B, src, dst, dinv, E, 256);
  gcn_finalize<<<CDIV((size_t)N * 256, TPB), TPB, 0, stream>>>(B, A, dinv, b1, (size_t)N * 256, 256);
  run_bn_act(B, stats, g1, be1, N, 256, stream);

  // ---- Layer 2: GAT 256 -> 128 (4 heads, mean) ----
  launch_gemm(B, gat_w, A, N, 256, 512, Xb, Wt, stream);
  gat_attn_coef<<<CDIV(N * 4, TPB), TPB, 0, stream>>>(A, gat_as, gat_ad, asc, adc, eself, mxu, N);
  gat_edge_max<<<CDIV(E * 4, (long long)TPB), TPB, 0, stream>>>(src, dst, asc, adc, mxu, E);
  gat_node_mid<<<CDIV(N * 4, TPB), TPB, 0, stream>>>(mxu, mxf, eself, denom, N);
  gat_edge_exp<<<CDIV(E * 4, (long long)TPB), TPB, 0, stream>>>(src, dst, asc, adc, mxf, ee, denom, E);
  k_fill<<<CDIV((size_t)N * 512, TPB), TPB, 0, stream>>>(B, (size_t)N * 512, 0.0f);
  gat_edge_scatter<<<CDIV(E * 512, (long long)TPB), TPB, 0, stream>>>(src, dst, A, ee, B, E);
  gat_finalize<<<CDIV(N * 128, TPB), TPB, 0, stream>>>(B, A, eself, denom, gat_b, C, N);
  run_bn_act(C, stats, g2, be2, N, 128, stream);

  // ---- Layer 3: GCN 128 -> 64 ----
  launch_gemm(C, w3, A, N, 128, 64, Xb, Wt, stream);
  k_fill<<<CDIV((size_t)N * 64, TPB), TPB, 0, stream>>>(B, (size_t)N * 64, 0.0f);
  gcn_scatter<<<CDIV(E * 64, (long long)TPB), TPB, 0, stream>>>(A, B, src, dst, dinv, E, 64);
  gcn_finalize<<<CDIV((size_t)N * 64, TPB), TPB, 0, stream>>>(B, A, dinv, b3, (size_t)N * 64, 64);
  run_bn_act(B, stats, g3, be3, N, 64, stream);

  // ---- Layer 4: GCN 64 -> 32 ----
  launch_gemm(B, w4, A, N, 64, 32, Xb, Wt, stream);
  k_fill<<<CDIV((size_t)N * 32, TPB), TPB, 0, stream>>>(C, (size_t)N * 32, 0.0f);
  gcn_scatter<<<CDIV(E * 32, (long long)TPB), TPB, 0, stream>>>(A, C, src, dst, dinv, E, 32);
  gcn_finalize<<<CDIV((size_t)N * 32, TPB), TPB, 0, stream>>>(C, A, dinv, b4, (size_t)N * 32, 32);
  run_bn_act(C, stats, g4, be4, N, 32, stream);

  // ---- Layer 5: GCN 32 -> 16 ----
  launch_gemm(C, w5, A, N, 32, 16, Xb, Wt, stream);
  k_fill<<<CDIV((size_t)N * 16, TPB), TPB, 0, stream>>>(B, (size_t)N * 16, 0.0f);
  gcn_scatter<<<CDIV(E * 16, (long long)TPB), TPB, 0, stream>>>(A, B, src, dst, dinv, E, 16);
  gcn_finalize<<<CDIV((size_t)N * 16, TPB), TPB, 0, stream>>>(B, A, dinv, b5, (size_t)N * 16, 16);
  run_bn_act(B, stats, g5, be5, N, 16, stream);

  // ---- MLP head + log_softmax ----
  head_kernel<<<CDIV(N, TPB), TPB, 0, stream>>>(B, fw1, fb1, fw2, fb2, (float*)d_out, N);
}
